// Attention_3393024164286
// MI455X (gfx1250) — matrix operand order repaired and checked by reading, not compiled, not run on hardware
//
#include <hip/hip_runtime.h>
#include <hip/hip_bf16.h>

typedef __attribute__((ext_vector_type(16))) __bf16 v16bf;
typedef __attribute__((ext_vector_type(8)))  __bf16 v8bf;
typedef __attribute__((ext_vector_type(4)))  __bf16 v4bf;
typedef __attribute__((ext_vector_type(8)))  float  v8f;
typedef __attribute__((ext_vector_type(4)))  float  v4f;

union bfrag16 { v16bf f; v8bf h[2]; };

#define DIM   1024
#define NSEQ  2048
#define NB    2
#define HEADS 16
#define DHEAD 64
#define HEAD_ELEMS ((size_t)NB * HEADS * NSEQ * DHEAD)   // 4,194,304 per Q/K/V

__device__ inline v4bf cvt4(v4f v) {
    v4bf o;
    o[0] = (__bf16)v[0]; o[1] = (__bf16)v[1];
    o[2] = (__bf16)v[2]; o[3] = (__bf16)v[3];
    return o;
}

// Fragment loaders. LDS row strides are multiples of 8 bf16 (16 B) so each
// v8bf access is a single aligned ds_load_b128.
__device__ inline v16bf load_a_frag(const __bf16* row, int half) {
    bfrag16 a;
    a.h[0] = *(const v8bf*)(row + 8 * half);
    a.h[1] = *(const v8bf*)(row + 16 + 8 * half);
    return a.f;
}
__device__ inline v16bf load_b_frag(const __bf16* nrow, int half) {
    bfrag16 b;
    b.h[0] = *(const v8bf*)(nrow + 8 * half);
    b.h[1] = *(const v8bf*)(nrow + 16 + 8 * half);
    return b.f;
}

// ---------------------------------------------------------------------------
// Kernel 1: qkv = x @ W_qkv (M=4096, K=1024, N=3072) -> bf16 Q/K/V [b,h,n,d].
// Block 256 thr (8 waves), tile 256(M) x 64(N), K-step 32.
// ---------------------------------------------------------------------------
__global__ __launch_bounds__(256) void qkv_gemm(const float* __restrict__ x,
                                                const float* __restrict__ Wqkv,
                                                __bf16* __restrict__ qkv) {
    __shared__ __bf16 As[256][40];   // [m][k], 32 k used
    __shared__ __bf16 Bt[64][40];    // [n][k] (transposed tile), 32 k used

    const int tid  = threadIdx.x;
    const int lane = tid & 31;
    const int wave = tid >> 5;
    const int half = lane >> 4;
    const int l16  = lane & 15;
    const int m0 = blockIdx.y * 256;
    const int n0 = blockIdx.x * 64;

    // per-thread staging bases (advance by constant per K-step)
    const v4f* xp = (const v4f*)x + (size_t)(m0 + (tid >> 3)) * (DIM / 4) + (tid & 7);
    const v4f* wp = (const v4f*)Wqkv + (size_t)(tid >> 4) * (3 * DIM / 4)
                    + (n0 >> 2) + (tid & 15);
    __bf16* asp = &As[tid >> 3][(tid & 7) * 4];
    const int ar = tid >> 4, ac = tid & 15;   // B staging coords

    v8f acc[2][4] = {};

    for (int k0 = 0; k0 < DIM; k0 += 32) {
        // ---- load phase: independent b128 loads (full MLP) ----
        v4f at[8];
        #pragma unroll
        for (int it = 0; it < 8; it++)
            at[it] = xp[(size_t)it * 32 * (DIM / 4)];
        v4f bt0 = wp[0];
        v4f bt1 = wp[(size_t)16 * (3 * DIM / 4)];
        xp += 8;                          // 32 f32 along K
        wp += (size_t)32 * (3 * DIM / 4); // 32 rows along K

        int kn = k0 + 32;
        if (kn < DIM) {
            __builtin_prefetch(&x[(size_t)(m0 + tid) * DIM + kn], 0, 0);
            if (tid < 64)
                __builtin_prefetch(&Wqkv[(size_t)(kn + (tid >> 1)) * (3 * DIM)
                                         + n0 + (tid & 1) * 32], 0, 0);
        }

        // ---- store phase ----
        #pragma unroll
        for (int it = 0; it < 8; it++)
            *(v4bf*)(asp + it * 32 * 40) = cvt4(at[it]);
        #pragma unroll
        for (int e = 0; e < 4; e++) {
            Bt[ac * 4 + e][ar]      = (__bf16)bt0[e];
            Bt[ac * 4 + e][ar + 16] = (__bf16)bt1[e];
        }
        __syncthreads();

        v16bf afrag[2];
        #pragma unroll
        for (int mt = 0; mt < 2; mt++)
            afrag[mt] = load_a_frag(&As[wave * 32 + mt * 16 + l16][0], half);
        #pragma unroll
        for (int nt = 0; nt < 4; nt++) {
            v16bf bfrag = load_b_frag(&Bt[nt * 16 + l16][0], half);
            #pragma unroll
            for (int mt = 0; mt < 2; mt++)
                acc[mt][nt] = __builtin_amdgcn_wmma_f32_16x16x32_bf16(
                    false, afrag[mt], false, bfrag, (short)0, acc[mt][nt], false, false);
        }
        __syncthreads();
    }

    #pragma unroll
    for (int mt = 0; mt < 2; mt++)
        #pragma unroll
        for (int nt = 0; nt < 4; nt++)
            #pragma unroll
            for (int r = 0; r < 8; r++) {
                int m = m0 + wave * 32 + mt * 16 + r + 8 * half;
                int n = n0 + nt * 16 + l16;
                int which = n >> 10;
                int rem   = n & 1023;
                int h = rem >> 6;
                int d = rem & 63;
                int bi = m >> 11;
                int ni = m & 2047;
                qkv[(size_t)which * HEAD_ELEMS +
                    (((size_t)(bi * HEADS + h) * NSEQ + ni) << 6) + d] =
                    (__bf16)acc[mt][nt][r];
            }
}

// ---------------------------------------------------------------------------
// Kernel 2: flash attention per (b*h). Block = 128 query rows, 8 waves.
// ---------------------------------------------------------------------------
__global__ __launch_bounds__(256) void flash_attn(const __bf16* __restrict__ Q,
                                                  const __bf16* __restrict__ K,
                                                  const __bf16* __restrict__ V,
                                                  __bf16* __restrict__ AO) {
    __shared__ __bf16 Qs[128][72];    // [qrow][d]
    __shared__ __bf16 Ks[64][72];     // [key][d]
    __shared__ __bf16 Vst[64][72];    // [d][key]   (transposed)
    __shared__ __bf16 Ps[8][16][72];  // per-wave P tile [m][key]

    const int tid  = threadIdx.x;
    const int lane = tid & 31;
    const int wave = tid >> 5;
    const int half = lane >> 4;
    const int l16  = lane & 15;
    const int bh = blockIdx.y;
    const int q0 = blockIdx.x * 128;
    const __bf16* Qb = Q + (size_t)bh * NSEQ * DHEAD;
    const __bf16* Kb = K + (size_t)bh * NSEQ * DHEAD;
    const __bf16* Vb = V + (size_t)bh * NSEQ * DHEAD;

    // Q tile: 128x64 bf16 (load phase then store phase)
    {
        const int qr = tid >> 3, qc = (tid & 7) * 8;
        v8bf qt[4];
        #pragma unroll
        for (int it = 0; it < 4; it++)
            qt[it] = *(const v8bf*)&Qb[(size_t)(q0 + qr + it * 32) * DHEAD + qc];
        #pragma unroll
        for (int it = 0; it < 4; it++)
            *(v8bf*)&Qs[qr + it * 32][qc] = qt[it];
    }
    __syncthreads();

    v16bf qa[2];
    #pragma unroll
    for (int ks2 = 0; ks2 < 2; ks2++)
        qa[ks2] = load_a_frag(&Qs[wave * 16 + l16][32 * ks2], half);

    v8f o[4] = {};
    float mrow[8], lrow[8];
    #pragma unroll
    for (int r = 0; r < 8; r++) { mrow[r] = -1e30f; lrow[r] = 0.f; }
    const float scale = 0.03125f;   // 1024^-0.5

    const int sr = tid >> 3, sc = (tid & 7) * 8;   // staging coords
    const __bf16* kp = Kb + (size_t)sr * DHEAD + sc;
    const __bf16* vp = Vb + (size_t)sr * DHEAD + sc;

    for (int j = 0; j < NSEQ; j += 64) {
        // ---- load phase: 4 independent b128 loads ----
        v8bf k0v = *(const v8bf*)(kp);
        v8bf k1v = *(const v8bf*)(kp + (size_t)32 * DHEAD);
        v8bf v0v = *(const v8bf*)(vp);
        v8bf v1v = *(const v8bf*)(vp + (size_t)32 * DHEAD);
        kp += (size_t)64 * DHEAD;
        vp += (size_t)64 * DHEAD;

        int jn = j + 64;
        if (jn < NSEQ) {
            if (tid < 64)
                __builtin_prefetch(&Kb[(size_t)(jn + tid) * DHEAD], 0, 0);
            else if (tid < 128)
                __builtin_prefetch(&Vb[(size_t)(jn + tid - 64) * DHEAD], 0, 0);
        }

        // ---- store phase ----
        *(v8bf*)&Ks[sr][sc]      = k0v;
        *(v8bf*)&Ks[sr + 32][sc] = k1v;
        #pragma unroll
        for (int e = 0; e < 8; e++) {
            Vst[sc + e][sr]      = v0v[e];
            Vst[sc + e][sr + 32] = v1v[e];
        }
        __syncthreads();

        // S = Q K^T  (wave strip: 16 x 64)
        v8f s[4] = {};
        #pragma unroll
        for (int nt = 0; nt < 4; nt++) {
            #pragma unroll
            for (int ks2 = 0; ks2 < 2; ks2++) {
                v16bf bfrag = load_b_frag(&Ks[nt * 16 + l16][32 * ks2], half);
                s[nt] = __builtin_amdgcn_wmma_f32_16x16x32_bf16(
                    false, qa[ks2], false, bfrag, (short)0, s[nt], false, false);
            }
        }

        // online softmax per owned row (row = wave*16 + r + 8*half)
        #pragma unroll
        for (int r = 0; r < 8; r++) {
            float mx = fmaxf(fmaxf(s[0][r], s[1][r]), fmaxf(s[2][r], s[3][r]));
            #pragma unroll
            for (int off = 8; off >= 1; off >>= 1)
                mx = fmaxf(mx, __shfl_xor(mx, off, 32));
            float mnew  = fmaxf(mrow[r], mx * scale);
            float alpha = __expf(mrow[r] - mnew);
            float rs = 0.f;
            #pragma unroll
            for (int nt = 0; nt < 4; nt++) {
                float p = __expf(s[nt][r] * scale - mnew);
                s[nt][r] = p;
                rs += p;
            }
            #pragma unroll
            for (int off = 8; off >= 1; off >>= 1)
                rs += __shfl_xor(rs, off, 32);
            lrow[r] = lrow[r] * alpha + rs;
            mrow[r] = mnew;
            #pragma unroll
            for (int dt = 0; dt < 4; dt++) o[dt][r] *= alpha;
        }

        // bounce P through per-wave LDS (C-layout -> A-layout)
        #pragma unroll
        for (int nt = 0; nt < 4; nt++)
            #pragma unroll
            for (int r = 0; r < 8; r++)
                Ps[wave][r + 8 * half][nt * 16 + l16] = (__bf16)s[nt][r];

        // O += P V
        #pragma unroll
        for (int dt = 0; dt < 4; dt++) {
            #pragma unroll
            for (int ks2 = 0; ks2 < 2; ks2++) {
                v16bf pa    = load_a_frag(&Ps[wave][l16][32 * ks2], half);
                v16bf bfrag = load_b_frag(&Vst[dt * 16 + l16][32 * ks2], half);
                o[dt] = __builtin_amdgcn_wmma_f32_16x16x32_bf16(
                    false, pa, false, bfrag, (short)0, o[dt], false, false);
            }
        }
        __syncthreads();
    }

    const int bi = bh >> 4, h = bh & 15;
    #pragma unroll
    for (int dt = 0; dt < 4; dt++)
        #pragma unroll
        for (int r = 0; r < 8; r++) {
            int row = q0 + wave * 16 + r + 8 * half;
            int d   = dt * 16 + l16;
            float val = o[dt][r] / lrow[r];
            AO[(((size_t)(bi * NSEQ + row)) << 10) + h * DHEAD + d] = (__bf16)val;
        }
}

// ---------------------------------------------------------------------------
// Kernel 3: out = AO @ W_out + b_out  (M=4096, K=1024, N=1024), f32 out.
// ---------------------------------------------------------------------------
__global__ __launch_bounds__(256) void out_proj(const __bf16* __restrict__ A,
                                                const float* __restrict__ W,
                                                const float* __restrict__ bias,
                                                float* __restrict__ out) {
    __shared__ __bf16 As[256][40];
    __shared__ __bf16 Bt[64][40];

    const int tid  = threadIdx.x;
    const int lane = tid & 31;
    const int wave = tid >> 5;
    const int half = lane >> 4;
    const int l16  = lane & 15;
    const int m0 = blockIdx.y * 256;
    const int n0 = blockIdx.x * 64;

    const __bf16* ap = A + (size_t)(m0 + (tid >> 2)) * DIM + (tid & 3) * 8;
    const v4f* wp = (const v4f*)W + (size_t)(tid >> 4) * (DIM / 4)
                    + (n0 >> 2) + (tid & 15);
    __bf16* asp = &As[tid >> 2][(tid & 3) * 8];
    const int ar = tid >> 4, ac = tid & 15;

    v8f acc[2][4] = {};

    for (int k0 = 0; k0 < DIM; k0 += 32) {
        // ---- load phase ----
        v8bf at[4];
        #pragma unroll
        for (int it = 0; it < 4; it++)
            at[it] = *(const v8bf*)(ap + (size_t)it * 64 * DIM);
        v4f bt0 = wp[0];
        v4f bt1 = wp[(size_t)16 * (DIM / 4)];
        ap += 32;
        wp += (size_t)32 * (DIM / 4);

        int kn = k0 + 32;
        if (kn < DIM) {
            if (tid < 128)
                __builtin_prefetch(&A[(size_t)(m0 + tid * 2) * DIM + kn], 0, 0);
            else
                __builtin_prefetch(&W[(size_t)(kn + ((tid - 128) >> 1)) * DIM
                                      + n0 + (tid & 1) * 32], 0, 0);
        }

        // ---- store phase ----
        #pragma unroll
        for (int it = 0; it < 4; it++)
            *(v8bf*)(asp + it * 64 * 40) = at[it];
        #pragma unroll
        for (int e = 0; e < 4; e++) {
            Bt[ac * 4 + e][ar]      = (__bf16)bt0[e];
            Bt[ac * 4 + e][ar + 16] = (__bf16)bt1[e];
        }
        __syncthreads();

        v16bf afrag[2];
        #pragma unroll
        for (int mt = 0; mt < 2; mt++)
            afrag[mt] = load_a_frag(&As[wave * 32 + mt * 16 + l16][0], half);
        #pragma unroll
        for (int nt = 0; nt < 4; nt++) {
            v16bf bfrag = load_b_frag(&Bt[nt * 16 + l16][0], half);
            #pragma unroll
            for (int mt = 0; mt < 2; mt++)
                acc[mt][nt] = __builtin_amdgcn_wmma_f32_16x16x32_bf16(
                    false, afrag[mt], false, bfrag, (short)0, acc[mt][nt], false, false);
        }
        __syncthreads();
    }

    #pragma unroll
    for (int mt = 0; mt < 2; mt++)
        #pragma unroll
        for (int nt = 0; nt < 4; nt++)
            #pragma unroll
            for (int r = 0; r < 8; r++) {
                int m = m0 + wave * 32 + mt * 16 + r + 8 * half;
                int n = n0 + nt * 16 + l16;
                out[(size_t)m * DIM + n] = acc[mt][nt][r] + bias[n];
            }
}

// ---------------------------------------------------------------------------
// Launch: x, W_qkv, W_out, b_out -> out. Workspace (32 MB):
//   Q | K | V bf16 [b,h,n,d], then attn-out bf16 [b,n,dim]
// ---------------------------------------------------------------------------
extern "C" void kernel_launch(void* const* d_in, const int* in_sizes, int n_in,
                              void* d_out, int out_size, void* d_ws, size_t ws_size,
                              hipStream_t stream) {
    (void)in_sizes; (void)n_in; (void)out_size; (void)ws_size;
    const float* x    = (const float*)d_in[0];
    const float* Wqkv = (const float*)d_in[1];
    const float* Wout = (const float*)d_in[2];
    const float* bout = (const float*)d_in[3];
    float* out = (float*)d_out;

    __bf16* q  = (__bf16*)d_ws;
    __bf16* k  = q + HEAD_ELEMS;
    __bf16* v  = k + HEAD_ELEMS;
    __bf16* ao = v + HEAD_ELEMS;

    dim3 blk(256);
    qkv_gemm <<<dim3(48, 16), blk, 0, stream>>>(x, Wqkv, q);
    flash_attn<<<dim3(16, 32), blk, 0, stream>>>(q, k, v, ao);
    out_proj <<<dim3(16, 16), blk, 0, stream>>>(ao, Wout, bout, out);
}
